// Stem_slic_23845658427412
// MI455X (gfx1250) — compile-verified, matrix operand order checked by reading, not run on hardware
//
#include <hip/hip_runtime.h>
#include <hip/hip_bf16.h>

typedef __attribute__((ext_vector_type(16))) _Float16 v16h;
typedef __attribute__((ext_vector_type(2)))  _Float16 h2;
typedef __attribute__((ext_vector_type(8)))  float    v8f;

#define BB 128
#define HH 224
#define WW 224
#define HW (HH * WW)
#define NSEG 196
#define MROWS (BB * NSEG)   // 25088 = 49 * 512

// ---------------------------------------------------------------------------
// Kernel 1: per-batch segment statistics via LDS atomics.
// counts / sum_x / sum_y are integer-exact; ch0 sums use float LDS atomics.
// stats layout: [B][NSEG][5] = {count, sum_x, sum_y, S1, S2}
// ---------------------------------------------------------------------------
__global__ __launch_bounds__(256)
void seg_stats_kernel(const float* __restrict__ x, const int* __restrict__ seg,
                      float* __restrict__ stats) {
    __shared__ int   sc[NSEG], sx[NSEG], sy[NSEG];
    __shared__ float s1[NSEG], s2[NSEG];
    const int b = blockIdx.x;
    const int t = threadIdx.x;
    for (int i = t; i < NSEG; i += blockDim.x) {
        sc[i] = 0; sx[i] = 0; sy[i] = 0; s1[i] = 0.0f; s2[i] = 0.0f;
    }
    __syncthreads();
    const float* ch0 = x + (size_t)b * 3 * HW;
    const int*   sm  = seg + (size_t)b * HW;
    for (int p = t; p < HW; p += blockDim.x) {
        int s  = sm[p];
        int xv = p % WW;
        int yv = p / WW;
        float v = ch0[p];
        atomicAdd(&sc[s], 1);
        atomicAdd(&sx[s], xv);
        atomicAdd(&sy[s], yv);
        atomicAdd(&s1[s], v);
        atomicAdd(&s2[s], v * v);
    }
    __syncthreads();
    for (int i = t; i < NSEG; i += blockDim.x) {
        float* o = stats + ((size_t)b * NSEG + i) * 5;
        o[0] = (float)sc[i];
        o[1] = (float)sx[i];
        o[2] = (float)sy[i];
        o[3] = s1[i];
        o[4] = s2[i];
    }
}

// ---------------------------------------------------------------------------
// Kernel 2: build the 11-channel feature vector per (b, segment), pack fp16
// row-major into A0 [MROWS, 32] (K zero-padded 11 -> 32 for WMMA K=32).
// ---------------------------------------------------------------------------
__global__ __launch_bounds__(256)
void feat_pack_kernel(const float* __restrict__ x, const float* __restrict__ stats,
                      _Float16* __restrict__ A0) {
    const int b = blockIdx.x;
    const int s = threadIdx.x;
    if (s >= NSEG) return;
    const float* st = stats + ((size_t)b * NSEG + s) * 5;
    const float cnt   = st[0];
    const bool  valid = cnt > 0.0f;
    const float denom = fmaxf(cnt, 1.0f);
    const float xc = valid ? st[1] / denom : 0.0f;
    const float yc = valid ? st[2] / denom : 0.0f;
    const float HWf  = (float)HW;
    const float mean = st[3] / HWf;
    const float var  = (st[4] - st[3] * st[3] / HWf) / (HWf - 1.0f);
    const float stdv = sqrtf(fmaxf(var, 1e-12f));
    float r = 0.0f, g = 0.0f, bl = 0.0f;
    if (valid) {
        int yr  = (int)rintf(yc);   // round-half-even, matches jnp.round
        int xr  = (int)rintf(xc);
        int off = yr * WW + xr;
        const float* xb = x + (size_t)b * 3 * HW;
        r  = xb[off];
        g  = xb[HW + off];
        bl = xb[2 * HW + off];
    }
    float f[11] = {xc, yc, mean, mean, mean, stdv, stdv, stdv, r, g, bl};
    _Float16* row = A0 + ((size_t)b * NSEG + s) * 32;
    #pragma unroll
    for (int i = 0; i < 11; ++i) row[i] = (_Float16)f[i];
    #pragma unroll
    for (int i = 11; i < 32; ++i) row[i] = (_Float16)0.0f;
}

// ---------------------------------------------------------------------------
// Kernel 3: fold BN into conv weights + bias and pre-swizzle the fp16 weight
// matrix into the CDNA5 WMMA B-fragment layout:
//   Bp[((kt*NT + nt)*32 + lane)*16 + 2*i + j] = W'[n, k]
//   n = nt*16 + (lane&15),  k = kt*32 + 8*(lane>>4) + (i<4 ? 2i : 16+2(i-4)) + j
// so each lane's 16 halves per K-tile are one contiguous 32-byte load.
// ---------------------------------------------------------------------------
__global__ __launch_bounds__(256)
void pack_weights_kernel(const float* __restrict__ W, const float* __restrict__ bias,
                         const float* __restrict__ gamma, const float* __restrict__ beta,
                         const float* __restrict__ rm, const float* __restrict__ rv,
                         int cin, int cout, int Kpad,
                         _Float16* __restrict__ Bp, float* __restrict__ bfold) {
    const int NT  = cout >> 4;
    const int KT  = Kpad >> 5;
    const int idx = blockIdx.x * blockDim.x + threadIdx.x;
    if (idx < cout) {
        float inv = gamma[idx] * rsqrtf(rv[idx] + 1e-5f);
        bfold[idx] = (bias[idx] - rm[idx]) * inv + beta[idx];
    }
    const int total = KT * NT * 32;
    if (idx >= total) return;
    const int lane = idx & 31;
    const int tile = idx >> 5;
    const int nt   = tile % NT;
    const int kt   = tile / NT;
    const int half = lane >> 4;
    const int n    = nt * 16 + (lane & 15);
    const float inv = gamma[n] * rsqrtf(rv[n] + 1e-5f);
    _Float16* o = Bp + (size_t)idx * 16;
    #pragma unroll
    for (int i = 0; i < 8; ++i) {
        int kb = (i < 4 ? 2 * i : 16 + 2 * (i - 4)) + 8 * half + kt * 32;
        #pragma unroll
        for (int j = 0; j < 2; ++j) {
            int k = kb + j;
            float v = (k < cin) ? W[(size_t)n * cin + k] * inv : 0.0f;
            o[2 * i + j] = (_Float16)v;
        }
    }
}

// ---------------------------------------------------------------------------
// Kernel 4: fused 1x1-conv GEMM via v_wmma_f32_16x16x32_f16 with 64x32
// register blocking per wave (4 M-subtiles x 2 N-subtiles, 8 accumulators).
// Block = 256 threads = 8 waves covering 512 rows; grid = (M/512, N/32).
// Per K-step: 4 A-fragment + 2 B-fragment loads feed 8 WMMAs (~22 FLOP/B).
// Epilogue: + folded bias, optional ReLU; intermediate -> fp16 row-major,
// final layer -> f32 transposed [b, n, s] into d_out.
// ---------------------------------------------------------------------------
__global__ __launch_bounds__(256)
void gemm_wmma_kernel(const _Float16* __restrict__ A, const _Float16* __restrict__ Bp,
                      const float* __restrict__ bias, int K, int N, int relu,
                      _Float16* __restrict__ outH, float* __restrict__ outF) {
    const int lane = threadIdx.x & 31;
    const int wave = threadIdx.x >> 5;
    const int row0 = blockIdx.x * 512 + wave * 64;   // 64 rows per wave
    const int nt0  = blockIdx.y * 2;                 // two 16-wide N tiles
    const int half = lane >> 4;
    const int l15  = lane & 15;

    v8f acc[4][2] = {};
    const int    KT      = K >> 5;
    const size_t bstride = (size_t)(N >> 4) * 512;   // halves per K-tile (NT*32*16)
    const _Float16* arow = A + (size_t)(row0 + l15) * K + 8 * half;
    const _Float16* bptr = Bp + ((size_t)nt0 * 32 + lane) * 16;

    for (int kt = 0; kt < KT; ++kt) {
        v16h b0 = *(const v16h*)(bptr);
        v16h b1 = *(const v16h*)(bptr + 512);        // next N tile: 32 lanes * 16 halves
        #pragma unroll
        for (int mi = 0; mi < 4; ++mi) {
            const _Float16* ap = arow + (size_t)(mi * 16) * K;
            v16h a;
            #pragma unroll
            for (int i = 0; i < 8; ++i) {
                int kb = (i < 4) ? 2 * i : 16 + 2 * (i - 4);  // +8*half folded into arow
                h2 p = *(const h2*)(ap + kb);
                a[2 * i]     = p[0];
                a[2 * i + 1] = p[1];
            }
            acc[mi][0] = __builtin_amdgcn_wmma_f32_16x16x32_f16(
                false, a, false, b0, (short)0, acc[mi][0], false, false);
            acc[mi][1] = __builtin_amdgcn_wmma_f32_16x16x32_f16(
                false, a, false, b1, (short)0, acc[mi][1], false, false);
        }
        arow += 32;
        bptr += bstride;
    }

    #pragma unroll
    for (int ni = 0; ni < 2; ++ni) {
        const int   n  = (nt0 + ni) * 16 + l15;      // C/D layout: col = lane&15
        const float bn = bias[n];
        #pragma unroll
        for (int mi = 0; mi < 4; ++mi) {
            #pragma unroll
            for (int r = 0; r < 8; ++r) {
                int   m = row0 + mi * 16 + r + 8 * half;  // row = r + 8*(lane>=16)
                float v = acc[mi][ni][r] + bn;
                if (relu) v = fmaxf(v, 0.0f);
                if (outH) {
                    outH[(size_t)m * N + n] = (_Float16)v;
                } else {
                    int b = m / NSEG;
                    int s = m - b * NSEG;
                    outF[((size_t)b * 768 + n) * NSEG + s] = v;
                }
            }
        }
    }
}

// ---------------------------------------------------------------------------
extern "C" void kernel_launch(void* const* d_in, const int* in_sizes, int n_in,
                              void* d_out, int out_size, void* d_ws, size_t ws_size,
                              hipStream_t stream) {
    (void)in_sizes; (void)n_in; (void)out_size; (void)ws_size;

    const float* x   = (const float*)d_in[0];
    const int*   seg = (const int*)d_in[1];
    // conv_params flattened: per layer (W, b, gamma, beta, rm, rv)
    const int cin_a [5] = {11, 96, 192, 384, 768};
    const int cout_a[5] = {96, 192, 384, 768, 768};
    const int kpad_a[5] = {32, 96, 192, 384, 768};

    // Output: (segmaps, conv_out) concatenated flat.
    hipMemcpyAsync(d_out, (const void*)seg, (size_t)BB * HW * sizeof(int),
                   hipMemcpyDeviceToDevice, stream);
    float* outConv = (float*)d_out + (size_t)BB * HW;

    // Workspace carve-up (256B aligned slices).
    char*  ws  = (char*)d_ws;
    size_t off = 0;
    auto take = [&](size_t bytes) -> char* {
        char* p = ws + off;
        off = (off + bytes + 255) & ~(size_t)255;
        return p;
    };
    float*    stats = (float*)   take((size_t)BB * NSEG * 5 * sizeof(float));
    _Float16* A0    = (_Float16*)take((size_t)MROWS * 32  * sizeof(_Float16));
    _Float16* bufA  = (_Float16*)take((size_t)MROWS * 384 * sizeof(_Float16)); // 96, 384
    _Float16* bufB  = (_Float16*)take((size_t)MROWS * 768 * sizeof(_Float16)); // 192, 768
    _Float16* Bp[5];
    float*    bf[5];
    for (int l = 0; l < 5; ++l)
        Bp[l] = (_Float16*)take((size_t)kpad_a[l] * cout_a[l] * sizeof(_Float16));
    for (int l = 0; l < 5; ++l)
        bf[l] = (float*)take((size_t)cout_a[l] * sizeof(float));

    // Stage 1: segment statistics + feature packing.
    seg_stats_kernel<<<BB, 256, 0, stream>>>(x, seg, stats);
    feat_pack_kernel<<<BB, 256, 0, stream>>>(x, stats, A0);

    // Stage 2: fold BN + pre-swizzle weights for each layer.
    for (int l = 0; l < 5; ++l) {
        const float* Wl = (const float*)d_in[2 + 6 * l + 0];
        const float* bl = (const float*)d_in[2 + 6 * l + 1];
        const float* gl = (const float*)d_in[2 + 6 * l + 2];
        const float* tl = (const float*)d_in[2 + 6 * l + 3];
        const float* ml = (const float*)d_in[2 + 6 * l + 4];
        const float* vl = (const float*)d_in[2 + 6 * l + 5];
        int lanes = (kpad_a[l] >> 5) * (cout_a[l] >> 4) * 32;
        int work  = lanes > cout_a[l] ? lanes : cout_a[l];
        pack_weights_kernel<<<(work + 255) / 256, 256, 0, stream>>>(
            Wl, bl, gl, tl, ml, vl, cin_a[l], cout_a[l], kpad_a[l], Bp[l], bf[l]);
    }

    // Stage 3: GEMM chain.  M = 25088 = 49 blocks of 512 rows (8 waves x 64).
    const _Float16* act[6] = {A0, bufA, bufB, bufA, bufB, nullptr};
    for (int l = 0; l < 5; ++l) {
        const int N = cout_a[l];
        dim3 grid(MROWS / 512, N / 32);
        _Float16* oh = (l < 4) ? (_Float16*)act[l + 1] : nullptr;
        float*    of = (l < 4) ? nullptr : outConv;
        gemm_wmma_kernel<<<grid, 256, 0, stream>>>(
            act[l], Bp[l], bf[l], kpad_a[l], N, /*relu=*/(l < 4) ? 1 : 0, oh, of);
    }
}
